// ASCNN_48421461295141
// MI455X (gfx1250) — matrix operand order repaired and checked by reading, not compiled, not run on hardware
//
#include <hip/hip_runtime.h>

// ---------------------------------------------------------------------------
// CDNA5 (gfx1250) ASCNN forward.  bf16 WMMA for all GEMM work, f32 accum.
// ---------------------------------------------------------------------------

#define B_  256
#define S_  256
#define E_  300
#define H_  300
#define C_  600       // 2H
#define G_  1200      // 4H
#define KC_ 608       // concat K = E+H padded to 19*32
#define CP_ 608       // channels padded to 38*16

typedef __attribute__((ext_vector_type(4)))  float  f32x4;
typedef __attribute__((ext_vector_type(8)))  float  v8f;
typedef __attribute__((ext_vector_type(16))) __bf16 v16bf;

union FragU { f32x4 q[2]; v16bf v; };

__device__ __forceinline__ __bf16 tobf(float f) {
    unsigned u = __builtin_bit_cast(unsigned, f);
    u += 0x7FFFu + ((u >> 16) & 1u);          // round-to-nearest-even
    unsigned short h = (unsigned short)(u >> 16);
    return __builtin_bit_cast(__bf16, h);
}
__device__ __forceinline__ __bf16 bfzero() {
    return __builtin_bit_cast(__bf16, (unsigned short)0);
}
__device__ __forceinline__ float sigm(float x) { return 1.0f / (1.0f + expf(-x)); }

// A-matrix fragment (16xK row-major, bf16) from `base` with row stride
// `stride` elems.  Lane layout: row M = lane&15; holds
// K = koff .. koff+15, koff = 16*(lane>>4).  Two aligned 16B loads.
__device__ __forceinline__ v16bf load_a_frag(const __bf16* base, int stride,
                                             int kbase, int lane) {
    const __bf16* row = base + (size_t)(lane & 15) * stride;
    int koff = (lane >> 4) * 16;
    FragU u;
    u.q[0] = *(const f32x4*)(row + kbase + koff);
    u.q[1] = *(const f32x4*)(row + kbase + koff + 8);
    return u.v;
}

// B-matrix fragment (Kx16 from a row-major [N][stride] weight array, i.e. the
// GEMM does A x W^T).  Lane: col N = nbase + (lane&15); holds
// K = kbase + 16*(lane>>4) .. +15  -> two aligned 16B loads.
__device__ __forceinline__ v16bf load_b_frag(const __bf16* W, int stride,
                                             int nbase, int kbase, int lane) {
    const __bf16* p = W + (size_t)(nbase + (lane & 15)) * stride
                        + kbase + ((lane >> 4) << 4);
    FragU u;
    u.q[0] = *(const f32x4*)(p);
    u.q[1] = *(const f32x4*)(p + 8);
    return u.v;
}

__device__ __forceinline__ v8f wmma_bf16(v16bf a, v16bf b, v8f c) {
    return __builtin_amdgcn_wmma_f32_16x16x32_bf16(
        /*neg_a=*/false, a, /*neg_b=*/false, b,
        /*c_mod=*/(short)0, c, /*reuse_a=*/false, /*reuse_b=*/false);
}

__device__ __forceinline__ float posw(int j, int left, int right, int L, float na) {
    if (j < left)   return 1.0f - (float)(left - j) / na;
    if (j <= right) return 0.0f;
    if (j < L)      return 1.0f - (float)(j - right) / na;
    return 0.0f;
}

// ---------------------------------------------------------------------------
// Prep kernels
// ---------------------------------------------------------------------------
__global__ __launch_bounds__(256) void len_kernel(
    const int* __restrict__ cat, const int* __restrict__ aspi,
    const int* __restrict__ lefti, int* __restrict__ Ls,
    int* __restrict__ asps, int* __restrict__ lefts) {
    int b = threadIdx.x;
    int L = 0, a = 0, l = 0;
    for (int s = 0; s < S_; ++s) L += (cat[(size_t)b * S_ + s] != 0);
    for (int s = 0; s < 16;  ++s) a += (aspi[(size_t)b * 16 + s] != 0);
    for (int s = 0; s < 128; ++s) l += (lefti[(size_t)b * 128 + s] != 0);
    Ls[b] = L; asps[b] = a; lefts[b] = l;
}

// Wcat[dir][g][k] bf16, k<300 -> W_ih[g][k]; 300<=k<600 -> W_hh[g][k-300]; pad 0
__global__ __launch_bounds__(256) void wcat_kernel(
    const float* __restrict__ WihF, const float* __restrict__ WhhF,
    const float* __restrict__ WihB, const float* __restrict__ WhhB,
    __bf16* __restrict__ Wcat) {
    int i = blockIdx.x * 256 + threadIdx.x;
    const int total = 2 * G_ * KC_;
    if (i >= total) return;
    int dir = i / (G_ * KC_);
    int rem = i - dir * (G_ * KC_);
    int g = rem / KC_, k = rem - g * KC_;
    const float* Wih = dir ? WihB : WihF;
    const float* Whh = dir ? WhhB : WhhF;
    float v = 0.0f;
    if (k < E_)           v = Wih[(size_t)g * E_ + k];
    else if (k < E_ + H_) v = Whh[(size_t)g * H_ + (k - E_)];
    Wcat[i] = tobf(v);
}

// cwb[layer][k][o][i] bf16 with o,i padded to 608. w layout [o][i][3].
__global__ __launch_bounds__(256) void wconv_kernel(
    const float* __restrict__ w1, const float* __restrict__ w2,
    __bf16* __restrict__ cwb) {
    int i = blockIdx.x * 256 + threadIdx.x;
    const int total = 2 * 3 * CP_ * CP_;
    if (i >= total) return;
    int layer = i / (3 * CP_ * CP_);
    int rem = i - layer * (3 * CP_ * CP_);
    int k = rem / (CP_ * CP_);
    int rem2 = rem - k * (CP_ * CP_);
    int o = rem2 / CP_, ii = rem2 - o * CP_;
    const float* w = layer ? w2 : w1;
    float v = 0.0f;
    if (o < C_ && ii < C_) v = w[((size_t)o * C_ + ii) * 3 + k];
    cwb[i] = tobf(v);
}

// ---------------------------------------------------------------------------
// Bidirectional LSTM.  grid = (B/16, 2dir), 256 threads (8 waves).
// Dynamic LDS: xh[16][608] bf16 | gates[16][1200] f32 | cst[16][304] f32
// ---------------------------------------------------------------------------
#define LSTM_LDS (16*KC_*2 + 16*G_*4 + 16*304*4)

__global__ __launch_bounds__(256) void lstm_kernel(
    const int* __restrict__ cat, const float* __restrict__ embed,
    const __bf16* __restrict__ Wcat, const float* __restrict__ b_f,
    const float* __restrict__ b_b, const int* __restrict__ Ls,
    float* __restrict__ text_out) {
    extern __shared__ char smem[];
    __bf16* xh   = (__bf16*)smem;                                 // [16][608]
    float* gates = (float*)(smem + 16 * KC_ * 2);                 // [16][1200]
    float* cst   = (float*)(smem + 16 * KC_ * 2 + 16 * G_ * 4);   // [16][304]
    __shared__ int Lsm[16];

    const int b0   = blockIdx.x * 16;
    const int dir  = blockIdx.y;
    const int tid  = threadIdx.x;
    const int lane = tid & 31;
    const int wave = tid >> 5;
    const float*  bias = dir ? b_b : b_f;
    const __bf16* W    = Wcat + (size_t)dir * G_ * KC_;

    if (tid < 16) Lsm[tid] = Ls[b0 + tid];
    for (int i = tid; i < 16 * 308; i += 256) {          // zero h-region + pad
        int m = i / 308, c = i - m * 308;
        xh[(size_t)m * KC_ + E_ + c] = bfzero();
    }
    for (int i = tid; i < 16 * 304; i += 256) cst[i] = 0.0f;
    __syncthreads();

    for (int t = 0; t < S_; ++t) {
        // ---- stage x_t (embedding gather -> bf16) --------------------------
        for (int i = tid; i < 16 * E_; i += 256) {
            int m = i / E_, e = i - m * E_;
            int Lm = Lsm[m];
            int pos = t;
            if (dir) pos = (t < Lm) ? (Lm - 1 - t) : t;
            int idx = cat[(size_t)(b0 + m) * S_ + pos];
            xh[(size_t)m * KC_ + e] = tobf(embed[(size_t)idx * E_ + e]);
        }
        __syncthreads();

        // ---- gates[16,1200] = [x;h] x Wcat^T + bias (WMMA bf16) -----------
        for (int nt = wave; nt < G_ / 16; nt += 8) {
            float bv = bias[nt * 16 + (lane & 15)];
            v8f acc = {bv, bv, bv, bv, bv, bv, bv, bv};
            for (int kt = 0; kt < KC_ / 32; ++kt) {
                v16bf a  = load_a_frag(xh, KC_, kt * 32, lane);
                v16bf bb = load_b_frag(W, KC_, nt * 16, kt * 32, lane);
                acc = wmma_bf16(a, bb, acc);
            }
            int n  = nt * 16 + (lane & 15);
            int m0 = (lane >> 4) * 8;
#pragma unroll
            for (int r = 0; r < 8; ++r)
                gates[(size_t)(m0 + r) * G_ + n] = acc[r];
        }
        __syncthreads();

        // ---- elementwise gate update, h -> LDS(bf16) + global(f32) --------
        for (int i = tid; i < 16 * H_; i += 256) {
            int m = i / H_, u = i - m * H_;
            const float* gm = gates + (size_t)m * G_;
            float gi = gm[u], gf = gm[H_ + u], gg = gm[2 * H_ + u], go = gm[3 * H_ + u];
            float c = sigm(gf) * cst[(size_t)m * 304 + u] + sigm(gi) * tanhf(gg);
            cst[(size_t)m * 304 + u] = c;
            float h = sigm(go) * tanhf(c);
            xh[(size_t)m * KC_ + E_ + u] = tobf(h);
            int Lm = Lsm[m];
            int pos = t;
            if (dir) pos = (t < Lm) ? (Lm - 1 - t) : t;
            float ho = (pos < Lm) ? h : 0.0f;
            text_out[((size_t)(b0 + m) * S_ + pos) * C_ + dir * H_ + u] = ho;
        }
        __syncthreads();
    }
}

// ---------------------------------------------------------------------------
// Windowed convs (WMMA) + collapsed attention + FC.  One block per batch row.
// ---------------------------------------------------------------------------
__global__ __launch_bounds__(256) void conv_attn_kernel(
    const float* __restrict__ text_out, const __bf16* __restrict__ cwb,
    const float* __restrict__ c1b, const float* __restrict__ c2b,
    const float* __restrict__ fcw, const float* __restrict__ fcb,
    const int* __restrict__ Ls, const int* __restrict__ asps,
    const int* __restrict__ lefts, float* __restrict__ out) {
    __shared__ __bf16 in1[18 * CP_];
    __shared__ __bf16 in2[18 * CP_];
    __shared__ float  vbuf[CP_];
    __shared__ float  sc[S_];
    __shared__ float  tmp[S_];
    __shared__ float  feat[C_];

    const int b = blockIdx.x;
    const int tid = threadIdx.x, lane = tid & 31, wave = tid >> 5;
    const int L = Ls[b], asp = asps[b], left = lefts[b];
    const int right = left + asp - 1;
    const float na = (float)(L - asp);
    const float* tob = text_out + (size_t)b * S_ * C_;

    // ---- stage conv1 input window (rows = pos left-2 .. left+13) ----------
    for (int i = tid; i < 18 * CP_; i += 256) {
        int r = i / CP_, c = i - r * CP_;
        int pos = left - 2 + r;
        float v = 0.0f;
        if (c < C_ && r < 16 && pos >= 0 && pos < S_)
            v = tob[(size_t)pos * C_ + c] * posw(pos, left, right, L, na);
        in1[i] = tobf(v);
        in2[i] = bfzero();
    }
    for (int i = tid; i < CP_; i += 256) vbuf[i] = 0.0f;
    __syncthreads();

    // ---- conv1: out row m <-> pos left-1+m ; A[m] = in1[m+k] --------------
    for (int nt = wave; nt < CP_ / 16; nt += 8) {
        int n = nt * 16 + (lane & 15);
        float bv = (n < C_) ? c1b[n] : 0.0f;
        v8f acc = {bv, bv, bv, bv, bv, bv, bv, bv};
        for (int k = 0; k < 3; ++k) {
            const __bf16* Wk = cwb + (size_t)k * CP_ * CP_;
            for (int kt = 0; kt < CP_ / 32; ++kt) {
                v16bf a  = load_a_frag(in1 + (size_t)k * CP_, CP_, kt * 32, lane);
                v16bf bb = load_b_frag(Wk, CP_, nt * 16, kt * 32, lane);
                acc = wmma_bf16(a, bb, acc);
            }
        }
        int m0 = (lane >> 4) * 8;
#pragma unroll
        for (int r = 0; r < 8; ++r) {
            int m = m0 + r, pos = left - 1 + m;
            if (n < C_ && m < asp + 2 && pos >= 0 && pos < S_) {
                float y = fmaxf(acc[r], 0.0f) * posw(pos, left, right, L, na);
                in2[(size_t)m * CP_ + n] = tobf(y);
            }
        }
    }
    __syncthreads();

    // ---- conv2 on pos left..right, ReLU, aspect-mask, sum over s -> vbuf --
    for (int nt = wave; nt < CP_ / 16; nt += 8) {
        int n = nt * 16 + (lane & 15);
        float bv = (n < C_) ? c2b[n] : 0.0f;
        v8f acc = {bv, bv, bv, bv, bv, bv, bv, bv};
        for (int k = 0; k < 3; ++k) {
            const __bf16* Wk = cwb + (size_t)(3 + k) * CP_ * CP_;
            for (int kt = 0; kt < CP_ / 32; ++kt) {
                v16bf a  = load_a_frag(in2 + (size_t)k * CP_, CP_, kt * 32, lane);
                v16bf bb = load_b_frag(Wk, CP_, nt * 16, kt * 32, lane);
                acc = wmma_bf16(a, bb, acc);
            }
        }
        int m0 = (lane >> 4) * 8;
        float part = 0.0f;
#pragma unroll
        for (int r = 0; r < 8; ++r) {
            int m = m0 + r;
            if (m < asp) part += fmaxf(acc[r], 0.0f);
        }
        if (n < C_ && part != 0.0f) atomicAdd(&vbuf[n], part);
    }
    __syncthreads();

    // ---- scores[t] = v . text_out[b,t,:] ----------------------------------
    {
        float s = 0.0f;
        const float* row = tob + (size_t)tid * C_;
        for (int c = 0; c < C_; ++c) s += vbuf[c] * row[c];
        sc[tid] = s;
        tmp[tid] = s;
    }
    __syncthreads();
    for (int off = 128; off > 0; off >>= 1) {
        if (tid < off) tmp[tid] = fmaxf(tmp[tid], tmp[tid + off]);
        __syncthreads();
    }
    float mx = tmp[0];
    __syncthreads();
    float e = expf(sc[tid] - mx);
    tmp[tid] = e;
    __syncthreads();
    for (int off = 128; off > 0; off >>= 1) {
        if (tid < off) tmp[tid] += tmp[tid + off];
        __syncthreads();
    }
    float inv = 1.0f / tmp[0];
    __syncthreads();
    sc[tid] = e * inv;          // alpha[t]
    __syncthreads();

    // ---- feat[c] = sum_t alpha[t] * text_out[b,t,c] -----------------------
    for (int c = tid; c < C_; c += 256) {
        float f = 0.0f;
        for (int t = 0; t < S_; ++t) f += sc[t] * tob[(size_t)t * C_ + c];
        feat[c] = f;
    }
    __syncthreads();

    // ---- FC (P=3) ---------------------------------------------------------
    if (tid < 3) {
        float s = fcb[tid];
        for (int c = 0; c < C_; ++c) s += feat[c] * fcw[(size_t)tid * C_ + c];
        out[(size_t)b * 3 + tid] = s;
    }
}

// ---------------------------------------------------------------------------
extern "C" void kernel_launch(void* const* d_in, const int* in_sizes, int n_in,
                              void* d_out, int out_size, void* d_ws, size_t ws_size,
                              hipStream_t stream) {
    const int*   cat    = (const int*)d_in[0];
    const int*   aspi   = (const int*)d_in[1];
    const int*   lefti  = (const int*)d_in[2];
    const float* embed  = (const float*)d_in[3];
    const float* WihF   = (const float*)d_in[4];
    const float* WhhF   = (const float*)d_in[5];
    const float* b_f    = (const float*)d_in[6];
    const float* WihB   = (const float*)d_in[7];
    const float* WhhB   = (const float*)d_in[8];
    const float* b_b    = (const float*)d_in[9];
    const float* w1     = (const float*)d_in[10];
    const float* c1b    = (const float*)d_in[11];
    const float* w2     = (const float*)d_in[12];
    const float* c2b    = (const float*)d_in[13];
    const float* fcw    = (const float*)d_in[14];
    const float* fcb    = (const float*)d_in[15];

    char* ws = (char*)d_ws;
    size_t off = 0;
    float*  text_out = (float*)(ws + off);  off += (size_t)B_ * S_ * C_ * 4;      // 157.3 MB
    __bf16* Wcat     = (__bf16*)(ws + off); off += (size_t)2 * G_ * KC_ * 2;      // 2.9 MB
    __bf16* cwb      = (__bf16*)(ws + off); off += (size_t)2 * 3 * CP_ * CP_ * 2; // 4.4 MB
    int* Ls    = (int*)(ws + off); off += B_ * 4;
    int* asps  = (int*)(ws + off); off += B_ * 4;
    int* lefts = (int*)(ws + off); off += B_ * 4;

    len_kernel<<<1, 256, 0, stream>>>(cat, aspi, lefti, Ls, asps, lefts);

    {
        int total = 2 * G_ * KC_;
        wcat_kernel<<<(total + 255) / 256, 256, 0, stream>>>(WihF, WhhF, WihB, WhhB, Wcat);
    }
    {
        int total = 2 * 3 * CP_ * CP_;
        wconv_kernel<<<(total + 255) / 256, 256, 0, stream>>>(w1, w2, cwb);
    }

    lstm_kernel<<<dim3(B_ / 16, 2), 256, LSTM_LDS, stream>>>(
        cat, embed, Wcat, b_f, b_b, Ls, text_out);

    conv_attn_kernel<<<B_, 256, 0, stream>>>(
        text_out, cwb, c1b, c2b, fcw, fcb, Ls, asps, lefts, (float*)d_out);
}
